// GCNConv_39041252720968
// MI455X (gfx1250) — matrix operand order repaired and checked by reading, not compiled, not run on hardware
//
#include <hip/hip_runtime.h>

typedef __attribute__((ext_vector_type(16))) _Float16 v16h;
typedef __attribute__((ext_vector_type(8)))  _Float16 v8h;
typedef __attribute__((ext_vector_type(8)))  float    v8f;
typedef __attribute__((ext_vector_type(4)))  _Float16 h4;

#define D 128              // D_IN == D_OUT == 128
#define LDS_STRIDE 136     // 128 + 8 halves: keeps 16B alignment, spreads banks

// ---------------------------------------------------------------------------
// Kernel 1: X' = X @ W   (fp16 in, fp32 accumulate via WMMA, fp16 out)
// One wave per 16-row tile; 8 waves / block; W transposed into LDS once/block.
// Flattened 32-step WMMA loop with explicit B double-buffering: B(i+1) loads
// issue before wmma(i) so ds latency hides under the matrix op.
// ---------------------------------------------------------------------------
__global__ __launch_bounds__(256) void gcn_gemm_wmma(
    const _Float16* __restrict__ X,   // [N, 128] row-major
    const _Float16* __restrict__ W,   // [128, 128] row-major [k][n]
    _Float16* __restrict__ Xp,        // [N, 128] row-major (workspace)
    int nRows, int nTiles) {
  __shared__ _Float16 wt[D * LDS_STRIDE];  // wt[n * LDS_STRIDE + k] = W[k][n]

  // Cooperative transpose of W into LDS (coalesced global reads).
  for (int i = threadIdx.x; i < D * D; i += 256) {
    int k = i >> 7;          // i / 128
    int n = i & (D - 1);     // i % 128
    wt[n * LDS_STRIDE + k] = W[i];
  }
  __syncthreads();

  const int lane = threadIdx.x & 31;
  // tile is wave-uniform: keep it in an SGPR (scalar tail branch, scalar addr).
  const int tile =
      __builtin_amdgcn_readfirstlane(blockIdx.x * 8 + (threadIdx.x >> 5));
  if (tile >= nTiles) return;

  const int rowlo = lane & 15;   // M (for A/C) or N (for B) within tile
  const int hi    = lane >> 4;   // half-wave select
  const bool fullTile = (tile * 16 + 16) <= nRows;  // SGPR compare

  // Row this lane sources for the A fragments (clamped for a partial tail tile).
  int arow = tile * 16 + rowlo;
  if (arow >= nRows) arow = nRows - 1;
  const _Float16* xr = X + (long)arow * D;

  // Preload A fragments for all 4 k-steps.
  // Contiguous k order used here (matches B, not AMD's table): lane holds K = kb..kb+7 (halves 0..7) and
  // K = kb+8..kb+15 (halves 8..15), kb = ks*32 + (hi ? 16 : 0).
  v16h a[4];
#pragma unroll
  for (int ks = 0; ks < 4; ++ks) {
    const int kb = ks * 32 + hi * 16;
    v8h lo  = *(const v8h*)(xr + kb);
    v8h hi8 = *(const v8h*)(xr + kb + 8);
#pragma unroll
    for (int j = 0; j < 8; ++j) { a[ks][j] = lo[j]; a[ks][8 + j] = hi8[j]; }
  }

  // B fragment loader. Contiguous k order (same as A): lanes 0-15 hold K=0..15,
  // lanes 16-31 hold K=16..31 -> 16 contiguous halves of wt at fixed n.
  auto loadB = [&](int i) -> v16h {
    const int ks = i >> 3, nt = i & 7;
    const int kb = ks * 32 + hi * 16;
    const int n  = nt * 16 + rowlo;
    const v8h* bp = (const v8h*)(&wt[n * LDS_STRIDE + kb]);
    v8h b0 = bp[0], b1 = bp[1];
    v16h b;
#pragma unroll
    for (int j = 0; j < 8; ++j) { b[j] = b0[j]; b[8 + j] = b1[j]; }
    return b;
  };

  // 8 independent accumulators, one per 16-column n-tile.
  v8f acc[8];
#pragma unroll
  for (int nt = 0; nt < 8; ++nt) acc[nt] = (v8f){};

  // Software-pipelined: i = ks*8 + nt.
  v16h bcur = loadB(0);
#pragma unroll
  for (int i = 0; i < 32; ++i) {
    v16h bnext = bcur;
    if (i < 31) bnext = loadB(i + 1);           // issue next B before wmma(i)
    acc[i & 7] = __builtin_amdgcn_wmma_f32_16x16x32_f16(
        /*neg_a=*/false, a[i >> 3], /*neg_b=*/false, bcur,
        /*c_mod=*/(short)0, acc[i & 7], /*reuse_a=*/false, /*reuse_b=*/false);
    bcur = bnext;
  }

  // Epilogue. C layout: VGPR v, lane L -> (m = v + 8*hi, n = rowlo) of tile.
  _Float16* xpr = Xp + (long)tile * 16 * D;
  if (fullTile) {
#pragma unroll
    for (int nt = 0; nt < 8; ++nt) {
#pragma unroll
      for (int v = 0; v < 8; ++v) {
        xpr[(long)(v + hi * 8) * D + nt * 16 + rowlo] = (_Float16)acc[nt][v];
      }
    }
  } else {
#pragma unroll
    for (int nt = 0; nt < 8; ++nt) {
#pragma unroll
      for (int v = 0; v < 8; ++v) {
        const int r = tile * 16 + v + hi * 8;
        if (r < nRows)
          xpr[(long)(v + hi * 8) * D + nt * 16 + rowlo] = (_Float16)acc[nt][v];
      }
    }
  }
}

// ---------------------------------------------------------------------------
// Kernel 2: CSR SpMM  out[i] = sum_e deg[e] * X'[col[e]]
// One wave per row; lane owns 4 output dims. row is forced uniform so edge
// metadata (col, deg) flows through the SCALAR pipe (SMEM + SALU), columns
// feed saddr-form gathers with a constant per-lane voffset, and per-edge
// VALU work is just the b64 gather + 4 mixed-precision FMAs.
// X' (25.6 MB) is L2-resident on MI455X (192 MB L2) -> gathers hit L2.
// ---------------------------------------------------------------------------
__global__ __launch_bounds__(256) void gcn_spmm(
    const _Float16* __restrict__ Xp,    // [N, 128]
    const int* __restrict__ rowp,       // [N+1]
    const int* __restrict__ colidx,     // [E]
    const _Float16* __restrict__ degs,  // [E]
    _Float16* __restrict__ out,         // [N, 128]
    int nRows) {
  const int lane = threadIdx.x & 31;
  const int row =
      __builtin_amdgcn_readfirstlane(blockIdx.x * 8 + (threadIdx.x >> 5));
  if (row >= nRows) return;

  const int start = rowp[row];   // uniform -> scalar load
  const int end   = rowp[row + 1];
  const int dbase = lane * 4;
  const _Float16* xpl = Xp + dbase;   // per-lane column offset folded once

  float acc0 = 0.f, acc1 = 0.f, acc2 = 0.f, acc3 = 0.f;

  for (int base = start; base < end; base += 32) {
    int n = end - base;
    if (n > 32) n = 32;

    if (n == 32 && (base & 1) == 0) {
      // Fast path: scalar-pipe edge metadata, 32 pipelined b64 gathers.
      const int* cp = colidx + base;                       // uniform
      const unsigned int* dp = (const unsigned int*)(degs + base);  // uniform
#pragma unroll
      for (int j2 = 0; j2 < 16; ++j2) {
        const unsigned int dd = dp[j2];                    // scalar: 2 degs
        const _Float16 dh0 =
            __builtin_bit_cast(_Float16, (unsigned short)(dd & 0xffffu));
        const _Float16 dh1 =
            __builtin_bit_cast(_Float16, (unsigned short)(dd >> 16));
        const float d0 = (float)dh0;
        const float d1 = (float)dh1;
        const int c0 = cp[2 * j2];                         // scalar column
        const int c1 = cp[2 * j2 + 1];
        h4 x0 = *(const h4*)(xpl + (long)c0 * D);          // 256B/row gather
        h4 x1 = *(const h4*)(xpl + (long)c1 * D);
        acc0 = fmaf((float)x0[0], d0, acc0);
        acc1 = fmaf((float)x0[1], d0, acc1);
        acc2 = fmaf((float)x0[2], d0, acc2);
        acc3 = fmaf((float)x0[3], d0, acc3);
        acc0 = fmaf((float)x1[0], d1, acc0);
        acc1 = fmaf((float)x1[1], d1, acc1);
        acc2 = fmaf((float)x1[2], d1, acc2);
        acc3 = fmaf((float)x1[3], d1, acc3);
      }
    } else {
      // Generic tail: lane-gather + v_readlane broadcast.
      int   colv = 0;
      float degv = 0.f;
      if (lane < n) {
        colv = colidx[base + lane];
        degv = (float)degs[base + lane];
      }
      const int degbits = __float_as_int(degv);
      for (int j = 0; j < n; ++j) {
        const int   c = __builtin_amdgcn_readlane(colv, j);
        const float d = __int_as_float(__builtin_amdgcn_readlane(degbits, j));
        h4 x = *(const h4*)(xpl + (long)c * D);
        acc0 = fmaf((float)x[0], d, acc0);
        acc1 = fmaf((float)x[1], d, acc1);
        acc2 = fmaf((float)x[2], d, acc2);
        acc3 = fmaf((float)x[3], d, acc3);
      }
    }
  }

  h4 o;
  o[0] = (_Float16)acc0;
  o[1] = (_Float16)acc1;
  o[2] = (_Float16)acc2;
  o[3] = (_Float16)acc3;
  // Output is streamed once -> nontemporal store, keep L2 for X'.
  __builtin_nontemporal_store(o, (h4*)(out + (long)row * D + dbase));
}

// ---------------------------------------------------------------------------
// Launch
// ---------------------------------------------------------------------------
extern "C" void kernel_launch(void* const* d_in, const int* in_sizes, int n_in,
                              void* d_out, int out_size, void* d_ws, size_t ws_size,
                              hipStream_t stream) {
  const _Float16* X    = (const _Float16*)d_in[0];  // [N,128] fp16
  const _Float16* W    = (const _Float16*)d_in[1];  // [128,128] fp16
  const int*      rowp = (const int*)d_in[2];       // [N+1]
  const int*      ci   = (const int*)d_in[3];       // [E]
  const _Float16* degs = (const _Float16*)d_in[4];  // [E] fp16
  _Float16*       out  = (_Float16*)d_out;          // [N,128] fp16
  _Float16*       Xp   = (_Float16*)d_ws;           // workspace [N,128] fp16

  const int nRows  = in_sizes[2] - 1;        // 100000
  const int nTiles = (nRows + 15) / 16;      // 6250

  dim3 g1((nTiles + 7) / 8);                 // 8 tiles (waves) per block
  gcn_gemm_wmma<<<g1, 256, 0, stream>>>(X, W, Xp, nRows, nTiles);

  dim3 g2((nRows + 7) / 8);                  // 8 rows (waves) per block
  gcn_spmm<<<g2, 256, 0, stream>>>(Xp, rowp, ci, degs, out, nRows);
}
